// NaiveAttention_41429254537804
// MI455X (gfx1250) — matrix operand order repaired and checked by reading, not compiled, not run on hardware
//
#include <hip/hip_runtime.h>
#include <hip/hip_bf16.h>

typedef __attribute__((ext_vector_type(16))) _Float16 v16h;
typedef __attribute__((ext_vector_type(8)))  _Float16 v8h;
typedef __attribute__((ext_vector_type(4)))  _Float16 v4h;
typedef __attribute__((ext_vector_type(8)))  float    v8f;

#define T_SEQ 2048
#define DHEAD 64
#define QTILE 128
#define KTILE 64
#define NWAVES 8
#define NTHREADS (NWAVES * 32)

// ---- DPP16 butterfly max across a 16-lane row (no LDS, no dscnt waits) ----
template <int CTRL>
__device__ __forceinline__ float fmax_dpp(float x) {
    int y = __builtin_amdgcn_mov_dpp(__float_as_int(x), CTRL, 0xF, 0xF, true);
    return fmaxf(x, __int_as_float(y));
}
__device__ __forceinline__ float rowmax16(float x) {
    x = fmax_dpp<0xB1>(x);   // quad_perm [1,0,3,2]  : xor 1
    x = fmax_dpp<0x4E>(x);   // quad_perm [2,3,0,1]  : xor 2
    x = fmax_dpp<0x141>(x);  // row_half_mirror      : xor 4
    x = fmax_dpp<0x140>(x);  // row_mirror           : xor 8
    return x;
}

// A-matrix fragment (16x32 f16), ISA layout: lanes 0-15 hold K={0..7,16..23},
// lanes 16-31 hold K={8..15,24..31}, 2 K-values packed per VGPR.
__device__ __forceinline__ v16h load_a_frag(const _Float16* rowp, int kchunk, int hs) {
    int dlow = kchunk * 32 + hs * 8;
    v8h lo = *(const v8h*)(rowp + dlow);        // ds_load_b128
    v8h hi = *(const v8h*)(rowp + dlow + 16);   // ds_load_b128
    v16h f;
#pragma unroll
    for (int i = 0; i < 8; ++i) { f[i] = lo[i]; f[i + 8] = hi[i]; }
    return f;
}

// B-matrix fragment (32x16 f16): lanes 0-15 hold K={0..7,16..23},
// lanes 16-31 hold K={8..15,24..31} (same order as the A fragment).
__device__ __forceinline__ v16h load_b_frag(const _Float16* rowp, int kchunk, int hs) {
    return load_a_frag(rowp, kchunk, hs);  // 2x ds_load_b128
}

__global__ void __launch_bounds__(NTHREADS)
fa_fwd_kernel(const float* __restrict__ Q, const float* __restrict__ K,
              const float* __restrict__ V, const int* __restrict__ causal_p,
              float* __restrict__ Out) {
    __shared__ _Float16 Qs[QTILE * DHEAD];         // [q_row][d]       16 KB
    __shared__ _Float16 Ks[KTILE * DHEAD];         // [key][d]          8 KB
    __shared__ _Float16 Vt[DHEAD * KTILE];         // [d][key]          8 KB
    __shared__ _Float16 Ps[NWAVES][16 * KTILE];    // per-wave P       16 KB

    const int qt   = blockIdx.x;
    const int bh   = blockIdx.y;
    const int tid  = threadIdx.x;
    const int wave = tid >> 5;
    const int lane = tid & 31;
    const int lh   = lane & 15;
    const int hs   = lane >> 4;

    const int causal = causal_p[0];
    const long base  = (long)bh * T_SEQ * DHEAD;
    const int  q0    = qt * QTILE;
    const int  wrow0 = q0 + wave * 16;     // first q row owned by this wave

    // ---- Stage Q tile -> LDS f16, pre-scaled by 1/sqrt(D) ----
    {
        const float scale = 0.125f;  // 1/sqrt(64)
#pragma unroll
        for (int it = 0; it < (QTILE * DHEAD / 4) / NTHREADS; ++it) {
            int idx = tid + it * NTHREADS;       // float4 slots
            int row = idx >> 4;
            int c4  = (idx & 15) << 2;
            float4 qv = *(const float4*)(Q + base + (long)(q0 + row) * DHEAD + c4);
            v4h h;
            h[0] = (_Float16)(qv.x * scale); h[1] = (_Float16)(qv.y * scale);
            h[2] = (_Float16)(qv.z * scale); h[3] = (_Float16)(qv.w * scale);
            *(v4h*)(&Qs[row * DHEAD + c4]) = h;
        }
    }
    __syncthreads();

    // Per-wave Q A-fragments, resident for the whole K loop
    const _Float16* qrow = &Qs[(wave * 16 + lh) * DHEAD];
    v16h aq0 = load_a_frag(qrow, 0, hs);
    v16h aq1 = load_a_frag(qrow, 1, hs);

    // Ones B-fragment for row-sum-by-WMMA (register splat, no LDS)
    v16h ones;
#pragma unroll
    for (int i = 0; i < 16; ++i) ones[i] = (_Float16)1.0f;

    v8f acc[4];
#pragma unroll
    for (int nd = 0; nd < 4; ++nd) { v8f z = {}; acc[nd] = z; }
    float mrow[8], lrow[8];
#pragma unroll
    for (int r = 0; r < 8; ++r) { mrow[r] = -__builtin_inff(); lrow[r] = 0.0f; }

    const int numK = T_SEQ / KTILE;
    const int jend = causal ? (QTILE / KTILE) * qt + (QTILE / KTILE - 1) : (numK - 1);

    for (int j = 0; j <= jend; ++j) {
        __syncthreads();  // previous tile fully consumed
        // ---- Stage K (row-major) and V (transposed) tiles -> LDS f16 ----
#pragma unroll
        for (int it = 0; it < (KTILE * DHEAD / 4) / NTHREADS; ++it) {
            int idx = tid + it * NTHREADS;
            int row = idx >> 4;
            int c4  = (idx & 15) << 2;
            long g  = base + (long)(j * KTILE + row) * DHEAD + c4;
            float4 kv = *(const float4*)(K + g);
            v4h h;
            h[0] = (_Float16)kv.x; h[1] = (_Float16)kv.y;
            h[2] = (_Float16)kv.z; h[3] = (_Float16)kv.w;
            *(v4h*)(&Ks[row * DHEAD + c4]) = h;
            float4 vv = *(const float4*)(V + g);
            Vt[(c4 + 0) * KTILE + row] = (_Float16)vv.x;
            Vt[(c4 + 1) * KTILE + row] = (_Float16)vv.y;
            Vt[(c4 + 2) * KTILE + row] = (_Float16)vv.z;
            Vt[(c4 + 3) * KTILE + row] = (_Float16)vv.w;
        }
        __syncthreads();

        // Prefetch next K/V tile while we compute (global_prefetch_b8)
        if (j < jend) {
            long gn = base + (long)((j + 1) * KTILE + (tid >> 2)) * DHEAD + (tid & 3) * 16;
            __builtin_prefetch(K + gn, 0, 1);
            __builtin_prefetch(V + gn, 0, 1);
        }

        // Causal: waves whose rows are all above this key tile have nothing to do.
        const bool active = (!causal) || (j * KTILE <= wrow0 + 15);
        if (active) {
            // ---- S = Q K^T : four 16x16 f32 fragments via WMMA ----
            v8f sfr[4];
#pragma unroll
            for (int n = 0; n < 4; ++n) {
                const _Float16* krow = &Ks[(n * 16 + lh) * DHEAD];
                v16h bk0 = load_b_frag(krow, 0, hs);
                v16h bk1 = load_b_frag(krow, 1, hs);
                v8f c = {};
                c = __builtin_amdgcn_wmma_f32_16x16x32_f16(false, aq0, false, bk0,
                                                           (short)0, c, false, false);
                c = __builtin_amdgcn_wmma_f32_16x16x32_f16(false, aq1, false, bk1,
                                                           (short)0, c, false, false);
                sfr[n] = c;
            }

            // ---- Causal mask (only tiles straddling the diagonal) ----
            if (causal && (j * KTILE + KTILE - 1 > wrow0)) {
#pragma unroll
                for (int n = 0; n < 4; ++n) {
                    int gk = j * KTILE + n * 16 + lh;
#pragma unroll
                    for (int r = 0; r < 8; ++r) {
                        int gq = wrow0 + r + 8 * hs;
                        if (gk > gq) sfr[n][r] = -__builtin_inff();
                    }
                }
            }

            // ---- Online softmax: DPP row-max, exp, rescale ----
            float alpha_r[8], nm_r[8];
#pragma unroll
            for (int r = 0; r < 8; ++r) {
                float rm = fmaxf(fmaxf(sfr[0][r], sfr[1][r]),
                                 fmaxf(sfr[2][r], sfr[3][r]));
                rm = rowmax16(rm);
                float nm = fmaxf(mrow[r], rm);
                nm_r[r] = nm;
                alpha_r[r] = __expf(mrow[r] - nm);
#pragma unroll
                for (int n = 0; n < 4; ++n)
                    sfr[n][r] = __expf(sfr[n][r] - nm);
            }

            // ---- C-layout -> A-layout for P via per-wave LDS (wave-ordered) ----
            _Float16* pw = &Ps[wave][0];
#pragma unroll
            for (int n = 0; n < 4; ++n)
#pragma unroll
                for (int r = 0; r < 8; ++r)
                    pw[(r + 8 * hs) * KTILE + n * 16 + lh] = (_Float16)sfr[n][r];

            const _Float16* prow = &pw[lh * KTILE];
            v16h ap0 = load_a_frag(prow, 0, hs);
            v16h ap1 = load_a_frag(prow, 1, hs);

            // ---- Row sums via WMMA against ones: crs[r] = sum_k P[r][k] ----
            v8f crs = {};
            crs = __builtin_amdgcn_wmma_f32_16x16x32_f16(false, ap0, false, ones,
                                                         (short)0, crs, false, false);
            crs = __builtin_amdgcn_wmma_f32_16x16x32_f16(false, ap1, false, ones,
                                                         (short)0, crs, false, false);

#pragma unroll
            for (int r = 0; r < 8; ++r) {
                lrow[r] = lrow[r] * alpha_r[r] + crs[r];
                mrow[r] = nm_r[r];
#pragma unroll
                for (int nd = 0; nd < 4; ++nd) acc[nd][r] *= alpha_r[r];
            }

            // ---- O += P V ----
#pragma unroll
            for (int nd = 0; nd < 4; ++nd) {
                const _Float16* vrow = &Vt[(nd * 16 + lh) * KTILE];
                v16h bv0 = load_b_frag(vrow, 0, hs);
                v16h bv1 = load_b_frag(vrow, 1, hs);
                acc[nd] = __builtin_amdgcn_wmma_f32_16x16x32_f16(
                    false, ap0, false, bv0, (short)0, acc[nd], false, false);
                acc[nd] = __builtin_amdgcn_wmma_f32_16x16x32_f16(
                    false, ap1, false, bv1, (short)0, acc[nd], false, false);
            }
        }
    }

    // ---- Normalize and write out (f32) ----
#pragma unroll
    for (int r = 0; r < 8; ++r) {
        float inv  = 1.0f / lrow[r];
        int   grow = wrow0 + r + 8 * hs;
#pragma unroll
        for (int nd = 0; nd < 4; ++nd)
            Out[base + (long)grow * DHEAD + nd * 16 + lh] = acc[nd][r] * inv;
    }
}

extern "C" void kernel_launch(void* const* d_in, const int* in_sizes, int n_in,
                              void* d_out, int out_size, void* d_ws, size_t ws_size,
                              hipStream_t stream) {
    const float* q = (const float*)d_in[0];
    const float* k = (const float*)d_in[1];
    const float* v = (const float*)d_in[2];
    const int* causal = (const int*)d_in[3];
    float* out = (float*)d_out;

    const int BH = in_sizes[0] / (T_SEQ * DHEAD);  // = B*H = 32
    dim3 grid(T_SEQ / QTILE, BH);
    fa_fwd_kernel<<<grid, NTHREADS, 0, stream>>>(q, k, v, causal, out);
}
